// ReSpropAttention_31576599560615
// MI455X (gfx1250) — compile-verified, matrix operand order checked by reading, not run on hardware
//
#include <hip/hip_runtime.h>
#include <hip/hip_bf16.h>

typedef __attribute__((ext_vector_type(16))) __bf16 v16bf;
typedef __attribute__((ext_vector_type(8)))  __bf16 v8bf;
typedef __attribute__((ext_vector_type(4)))  __bf16 v4bf;
typedef __attribute__((ext_vector_type(2)))  __bf16 v2bf;
typedef __attribute__((ext_vector_type(8)))  float  v8f;
typedef __attribute__((ext_vector_type(4)))  float  v4f;

#define BATCH   32
#define SEQ     2048
#define DIM     128
#define BLOCK_M 128     // queries per workgroup (8 waves x 16)
#define KTILE   32      // keys per inner iteration
#define NIT     (SEQ / KTILE)
#define KSTR    136     // ldsK row stride in bf16 (272B = 17*16B -> aligned b128, banks spread)
#define VSTR    40      // ldsVt row stride in bf16 (80B = 5*16B -> aligned b128)

#define SHUF16(lo, hi) __builtin_shufflevector(lo, hi, 0,1,2,3,4,5,6,7,8,9,10,11,12,13,14,15)

__global__ __launch_bounds__(256)
void resprop_attn_fwd(const float* __restrict__ Qg, const float* __restrict__ Kg,
                      const float* __restrict__ Vg, float* __restrict__ Og)
{
    // double-buffered staging tiles
    __shared__ __bf16 ldsK [2][KTILE * KSTR];   // K tile, row-major [k][d]
    __shared__ __bf16 ldsVt[2][DIM   * VSTR];   // V tile, transposed [d][k]

    const int tid  = threadIdx.x;
    const int lane = tid & 31;
    const int lo16 = lane & 15;
    const int hi   = lane >> 4;          // 0 or 1 (wave half)
    const int wave = tid >> 5;

    const int b  = blockIdx.y;
    const int q0 = blockIdx.x * BLOCK_M + wave * 16;

    const float* Qb = Qg + (size_t)b * SEQ * DIM;
    const float* Kb = Kg + (size_t)b * SEQ * DIM;
    const float* Vb = Vg + (size_t)b * SEQ * DIM;
    float*       Ob = Og + (size_t)b * SEQ * DIM;

    // fold 1/sqrt(128) and log2(e) into Q -> softmax in base 2
    const float qscale = 0.088388347648318447f * 1.4426950408889634f;

    // ---- staging address map (per thread) ----
    // K: 4 float4 chunks; chunk j: e4 = tid + 256*j in [0,1024): k = e4>>5, d = (e4&31)*4
    int kk[4], kd[4];
    #pragma unroll
    for (int j = 0; j < 4; ++j) { const int e4 = tid + 256 * j; kk[j] = e4 >> 5; kd[j] = (e4 & 31) << 2; }
    // V: 2 blocks of (k-pair x 4d); e = tid + 256*j in [0,512): k2 = e>>5, d = (e&31)*4
    int vk2[2], vd[2];
    #pragma unroll
    for (int j = 0; j < 2; ++j) { const int e = tid + 256 * j; vk2[j] = e >> 5; vd[j] = (e & 31) << 2; }

    // ---- Q tile -> registers, WMMA B-operand layout, pre-scaled bf16 ----
    // B-operand: lane holds column q = lane&15; element i <-> d = c*32 + hi*8 + i + (i>=8 ? 8 : 0)
    v16bf qreg[4];
    {
        const float* qp = Qb + (size_t)(q0 + lo16) * DIM;
        #pragma unroll
        for (int c = 0; c < 4; ++c) {
            const int d0 = c * 32 + hi * 8;
            v16bf qv;
            #pragma unroll
            for (int i = 0; i < 8; ++i) {
                qv[i]     = (__bf16)(qp[d0 + i]      * qscale);
                qv[i + 8] = (__bf16)(qp[d0 + 16 + i] * qscale);
            }
            qreg[c] = qv;
        }
    }

    const v8f vzero = {};
    v8f acc[8];                      // O tile: 16 q-rows x 128 d, f32
    #pragma unroll
    for (int t = 0; t < 8; ++t) acc[t] = vzero;

    float m_run = -1e30f;            // running max (per query, replicated on lane & lane+16)
    float l_run = 0.0f;              // running sum

    // staged-tile registers (f32, carried across the compute phase)
    v4f kreg[4], vreg0[2], vreg1[2];

    // ---- prologue: load + store tile 0 into buffer 0 ----
    #pragma unroll
    for (int j = 0; j < 4; ++j)
        kreg[j] = *(const v4f*)(Kb + (size_t)kk[j] * DIM + kd[j]);
    #pragma unroll
    for (int j = 0; j < 2; ++j) {
        vreg0[j] = *(const v4f*)(Vb + (size_t)(2 * vk2[j] + 0) * DIM + vd[j]);
        vreg1[j] = *(const v4f*)(Vb + (size_t)(2 * vk2[j] + 1) * DIM + vd[j]);
    }
    {
        __bf16* lk = ldsK[0]; __bf16* lv = ldsVt[0];
        #pragma unroll
        for (int j = 0; j < 4; ++j) {
            v4bf kh;
            kh[0] = (__bf16)kreg[j][0]; kh[1] = (__bf16)kreg[j][1];
            kh[2] = (__bf16)kreg[j][2]; kh[3] = (__bf16)kreg[j][3];
            *(v4bf*)&lk[kk[j] * KSTR + kd[j]] = kh;
        }
        #pragma unroll
        for (int j = 0; j < 2; ++j) {
            #pragma unroll
            for (int x = 0; x < 4; ++x) {
                v2bf h; h[0] = (__bf16)vreg0[j][x]; h[1] = (__bf16)vreg1[j][x];
                *(v2bf*)&lv[(vd[j] + x) * VSTR + 2 * vk2[j]] = h;
            }
        }
    }

    for (int it = 0; it < NIT; ++it) {
        const int kb  = it * KTILE;
        const int cur = it & 1;
        const int nxt = cur ^ 1;
        const bool have_next = (it + 1) < NIT;

        __syncthreads();   // buffer `cur` fully staged; previous reads of `nxt` done

        // ---- issue global loads for tile it+1 (latency hidden behind WMMAs) ----
        if (have_next) {
            const float* Kn = Kb + (size_t)(kb + KTILE) * DIM;
            const float* Vn = Vb + (size_t)(kb + KTILE) * DIM;
            #pragma unroll
            for (int j = 0; j < 4; ++j)
                kreg[j] = *(const v4f*)(Kn + (size_t)kk[j] * DIM + kd[j]);
            #pragma unroll
            for (int j = 0; j < 2; ++j) {
                vreg0[j] = *(const v4f*)(Vn + (size_t)(2 * vk2[j] + 0) * DIM + vd[j]);
                vreg1[j] = *(const v4f*)(Vn + (size_t)(2 * vk2[j] + 1) * DIM + vd[j]);
            }
        }
        // prefetch tile it+2 into L2/WGP$ (one 64B line per thread covers 16KB tile)
        if (kb + 2 * KTILE < SEQ) {
            __builtin_prefetch(Kb + (size_t)(kb + 2 * KTILE) * DIM + tid * 16, 0, 1);
            __builtin_prefetch(Vb + (size_t)(kb + 2 * KTILE) * DIM + tid * 16, 0, 1);
        }

        const __bf16* lk = ldsK[cur];
        const __bf16* lv = ldsVt[cur];

        // ---- S^T = K_tile . Q^T  (two 16x16 tiles: keys 0-15 and 16-31) ----
        v8f s0 = vzero, s1 = vzero;
        #pragma unroll
        for (int c = 0; c < 4; ++c) {
            const int d0 = c * 32 + hi * 8;
            const __bf16* kp0 = &lk[ lo16       * KSTR + d0];
            const __bf16* kp1 = &lk[(lo16 + 16) * KSTR + d0];
            const v16bf a0 = SHUF16(*(const v8bf*)kp0, *(const v8bf*)(kp0 + 16));
            const v16bf a1 = SHUF16(*(const v8bf*)kp1, *(const v8bf*)(kp1 + 16));
            s0 = __builtin_amdgcn_wmma_f32_16x16x32_bf16(false, a0, false, qreg[c],
                                                         (short)0, s0, false, false);
            s1 = __builtin_amdgcn_wmma_f32_16x16x32_bf16(false, a1, false, qreg[c],
                                                         (short)0, s1, false, false);
        }

        // ---- online softmax (base-2); lane holds query q = lane&15 ----
        float mt = -1e30f;
        #pragma unroll
        for (int i = 0; i < 8; ++i) { mt = fmaxf(mt, s0[i]); mt = fmaxf(mt, s1[i]); }
        mt = fmaxf(mt, __shfl_xor(mt, 16, 32));
        const float m_new = fmaxf(m_run, mt);
        const float alpha = __builtin_amdgcn_exp2f(m_run - m_new);

        float lsum = 0.0f;
        v8bf p0, p1;
        #pragma unroll
        for (int i = 0; i < 8; ++i) {
            const float e0 = __builtin_amdgcn_exp2f(s0[i] - m_new);
            const float e1 = __builtin_amdgcn_exp2f(s1[i] - m_new);
            lsum += e0 + e1;
            p0[i] = (__bf16)e0;
            p1[i] = (__bf16)e1;
        }
        lsum += __shfl_xor(lsum, 16, 32);
        l_run = l_run * alpha + lsum;
        m_run = m_new;

        // Two stacked S^T D-tiles == A-operand layout of P (16q x 32k): free repack.
        const v16bf pA = SHUF16(p0, p1);

        // alpha per accumulator row: O row q = r + hi*8; stats live on lane q (and q+16)
        float alpha_r[8];
        #pragma unroll
        for (int r = 0; r < 8; ++r) alpha_r[r] = __shfl(alpha, hi * 8 + r, 32);

        // ---- O = diag(alpha).O + P . V_tile ----
        #pragma unroll
        for (int t = 0; t < 8; ++t) {
            const __bf16* vp = &lv[(t * 16 + lo16) * VSTR + hi * 8];
            const v16bf vB = SHUF16(*(const v8bf*)vp, *(const v8bf*)(vp + 16));
            v8f o = acc[t];
            #pragma unroll
            for (int r = 0; r < 8; ++r) o[r] *= alpha_r[r];
            acc[t] = __builtin_amdgcn_wmma_f32_16x16x32_bf16(false, pA, false, vB,
                                                             (short)0, o, false, false);
        }

        // ---- convert + store tile it+1 into buffer `nxt` (after compute) ----
        if (have_next) {
            __bf16* lkn = ldsK[nxt]; __bf16* lvn = ldsVt[nxt];
            #pragma unroll
            for (int j = 0; j < 4; ++j) {
                v4bf kh;
                kh[0] = (__bf16)kreg[j][0]; kh[1] = (__bf16)kreg[j][1];
                kh[2] = (__bf16)kreg[j][2]; kh[3] = (__bf16)kreg[j][3];
                *(v4bf*)&lkn[kk[j] * KSTR + kd[j]] = kh;
            }
            #pragma unroll
            for (int j = 0; j < 2; ++j) {
                #pragma unroll
                for (int x = 0; x < 4; ++x) {
                    v2bf h; h[0] = (__bf16)vreg0[j][x]; h[1] = (__bf16)vreg1[j][x];
                    *(v2bf*)&lvn[(vd[j] + x) * VSTR + 2 * vk2[j]] = h;
                }
            }
        }
    }

    // ---- epilogue: divide by l, store f32 ----
    float inv_l[8];
    #pragma unroll
    for (int r = 0; r < 8; ++r) inv_l[r] = 1.0f / __shfl(l_run, hi * 8 + r, 32);

    #pragma unroll
    for (int t = 0; t < 8; ++t) {
        #pragma unroll
        for (int r = 0; r < 8; ++r) {
            Ob[(size_t)(q0 + r + hi * 8) * DIM + t * 16 + lo16] = acc[t][r] * inv_l[r];
        }
    }
}

extern "C" void kernel_launch(void* const* d_in, const int* in_sizes, int n_in,
                              void* d_out, int out_size, void* d_ws, size_t ws_size,
                              hipStream_t stream) {
    (void)in_sizes; (void)n_in; (void)out_size; (void)d_ws; (void)ws_size;
    const float* Q = (const float*)d_in[0];
    const float* K = (const float*)d_in[1];
    const float* V = (const float*)d_in[2];
    float* O = (float*)d_out;
    dim3 grid(SEQ / BLOCK_M, BATCH);
    resprop_attn_fwd<<<grid, 256, 0, stream>>>(Q, K, V, O);
}
